// EchoModel_75892072121097
// MI455X (gfx1250) — matrix operand order repaired and checked by reading, not compiled, not run on hardware
//
#include <hip/hip_runtime.h>
#include <hip/hip_bf16.h>
#include <math.h>

// Model constants
#define VC 50257
#define DC 128
#define LC 2
#define NC 512
#define HC 8
#define DHC 16
#define BC 2
#define SC 256
#define PHI_F 1.6180339887498949f
#define INV_SCALE 0.17677669529663687f  // 1/sqrt(2*DH) = 1/sqrt(32)

typedef __attribute__((ext_vector_type(16))) __bf16 v16bf;
typedef __attribute__((ext_vector_type(8)))  float  v8f;

// ---------------------------------------------------------------------------
// Kernel 0: embedding gather + sequential oscillator scan -> states, x
// (recurrence-sensitive: keep precise sincosf)
// ---------------------------------------------------------------------------
__global__ void scan_kernel(const int* __restrict__ tokens,
                            const float* __restrict__ emb,
                            float* __restrict__ states,
                            float* __restrict__ x) {
    int t = blockIdx.x * blockDim.x + threadIdx.x;
    if (t >= BC * DC) return;
    int b = t / DC, d = t % DC;
    float s = 0.0f;
    for (int i = 0; i < SC; ++i) {
        int tok = tokens[b * SC + i];
        float w  = emb[(size_t)tok * (2 * DC) + d];
        float bb = emb[(size_t)tok * (2 * DC) + DC + d];
        float theta = s / (1.0f + fabsf(w)) + bb + (float)i * PHI_F;
        float sn, cs;
        sincosf(theta, &sn, &cs);
        s = cs + sn;
        states[(b * SC + i) * DC + d] = s;
        x[(b * SC + i) * DC + d] = s;
    }
}

// ---------------------------------------------------------------------------
// Kernel 1: attention for one layer. One block per (b, q).
// ---------------------------------------------------------------------------
__global__ void attn_kernel(const float* __restrict__ x,
                            const float* __restrict__ states,
                            const float* __restrict__ wq, const float* __restrict__ bq,
                            const float* __restrict__ wk, const float* __restrict__ bk,
                            const float* __restrict__ w_out, const float* __restrict__ b_out,
                            const float* __restrict__ out_scale,
                            const float* __restrict__ attn_scale,
                            float* __restrict__ attn, int l) {
    int b = blockIdx.x / SC;
    int q = blockIdx.x % SC;
    int tid = threadIdx.x;

    __shared__ float sh_x[DC];
    __shared__ float sh_q[2 * DHC];
    __shared__ float sh_w2[SC];
    __shared__ float red[256];

    const float* wq_l = wq + l * DC;
    const float* bq_l = bq + l * DC;
    const float* wk_l = wk + l * DC;
    const float* bk_l = bk + l * DC;
    const float* wo_l = w_out + l * DC;
    const float* bo_l = b_out + l * DC;
    const float* os_l = out_scale + l * DC;
    float ascale = attn_scale[l];
    float qphi = (float)q * PHI_F;

    if (tid < DC) sh_x[tid] = x[(b * SC + q) * DC + tid];
    sh_w2[tid] = 0.0f;
    __syncthreads();

    for (int h = 0; h < HC; ++h) {
        if (tid < DHC) {
            int idx = h * DHC + tid;
            float th = sh_x[idx] / (1.0f + fabsf(wq_l[idx])) + bq_l[idx] + qphi;
            float sn, cs;
            sincosf(th, &sn, &cs);    // q has +q*PHI (large arg): precise
            sh_q[tid] = cs;
            sh_q[tid + DHC] = sn;
        }
        __syncthreads();

        const float* st = states + (b * SC + tid) * DC + h * DHC;
        float dot = 0.0f;
        #pragma unroll
        for (int j = 0; j < DHC; ++j) {
            int idx = h * DHC + j;
            float th = st[j] / (1.0f + fabsf(wk_l[idx])) + bk_l[idx];
            float sn, cs;
            __sincosf(th, &sn, &cs);  // small args: fast hw sincos
            dot += cs * sh_q[j] + sn * sh_q[j + DHC];
        }
        float score = dot * INV_SCALE;
        bool valid = (tid <= q);

        red[tid] = valid ? score : -3.402823466e38f;
        __syncthreads();
        for (int off = 128; off > 0; off >>= 1) {
            if (tid < off) red[tid] = fmaxf(red[tid], red[tid + off]);
            __syncthreads();
        }
        float m = red[0];
        __syncthreads();

        float e = valid ? expf(score - m) : 0.0f;
        red[tid] = e;
        __syncthreads();
        for (int off = 128; off > 0; off >>= 1) {
            if (tid < off) red[tid] += red[tid + off];
            __syncthreads();
        }
        float inv = 1.0f / red[0];
        __syncthreads();

        sh_w2[tid] += e * inv;
        __syncthreads();
    }

    if (tid < DC) {
        float c = 0.0f;
        for (int k = 0; k < SC; ++k)
            c += sh_w2[k] * states[(b * SC + k) * DC + tid];
        float th = c / (1.0f + fabsf(wo_l[tid])) + bo_l[tid] + qphi;
        float sn, cs;
        sincosf(th, &sn, &cs);
        attn[(b * SC + q) * DC + tid] = ascale * os_l[tid] * (cs + sn);
    }
}

// ---------------------------------------------------------------------------
// Kernel 2a: per-layer precompute invw = 1/(1+|Wf|) (hoists v_rcp out of the
// 33.5M-element fourier inner loop).
// ---------------------------------------------------------------------------
__global__ void invw_kernel(const float* __restrict__ Wf,
                            float* __restrict__ invw, int l) {
    int i = blockIdx.x * blockDim.x + threadIdx.x;
    if (i >= NC * DC) return;
    invw[i] = 1.0f / (1.0f + fabsf(Wf[(size_t)l * NC * DC + i]));
}

// ---------------------------------------------------------------------------
// Kernel 2b: fourier feature sums. One block per (b,s); thread strides n.
// Angle-addition: sum cos(u+t) = cos t * C - sin t * S, keeping u small so
// the fast hardware sincos path is valid; inner loop is fma + __sincosf.
// ---------------------------------------------------------------------------
__global__ void fourier_kernel(const float* __restrict__ x,
                               const float* __restrict__ invw,
                               const float* __restrict__ Bf,
                               float* __restrict__ cos_sum,
                               float* __restrict__ sin_sum, int l) {
    int bs = blockIdx.x;
    int s  = bs % SC;
    int tid = threadIdx.x;
    __shared__ float sh_x[DC];
    if (tid < DC) sh_x[tid] = x[bs * DC + tid];
    __syncthreads();

    const float* Bl = Bf + (size_t)l * NC * DC;
    float st_, ct_;
    sincosf((float)s, &st_, &ct_);           // once, precise (t up to 255)

    for (int n = tid; n < NC; n += blockDim.x) {
        const float* iw = invw + n * DC;
        const float* bb = Bl + n * DC;
        float C = 0.0f, S = 0.0f;
        #pragma unroll 4
        for (int d = 0; d < DC; ++d) {
            float u = __builtin_fmaf(sh_x[d], iw[d], bb[d]);
            float su, cu;
            __sincosf(u, &su, &cu);          // fast hw sincos, small args
            C += cu;
            S += su;
        }
        cos_sum[bs * NC + n] = ct_ * C - st_ * S;
        sin_sum[bs * NC + n] = st_ * C + ct_ * S;
    }
}

// ---------------------------------------------------------------------------
// Kernel 3: res GEMM + silu + residual combine. One block per (b,s).
// ---------------------------------------------------------------------------
__global__ void res_kernel(float* __restrict__ x,
                           const float* __restrict__ attn,
                           const float* __restrict__ cos_sum,
                           const float* __restrict__ sin_sum,
                           const float* __restrict__ proj_real,
                           const float* __restrict__ proj_imag,
                           const float* __restrict__ res_scale, int l) {
    int bs = blockIdx.x;
    int d  = threadIdx.x;     // blockDim == 128
    __shared__ float sc[NC];
    __shared__ float ssh[NC];
    for (int n = d; n < NC; n += blockDim.x) {
        sc[n]  = cos_sum[bs * NC + n];
        ssh[n] = sin_sum[bs * NC + n];
    }
    __syncthreads();

    const float* pr = proj_real + (size_t)l * DC * NC + (size_t)d * NC;
    const float* pi = proj_imag + (size_t)l * DC * NC + (size_t)d * NC;
    float acc = 0.0f;
    #pragma unroll 4
    for (int n = 0; n < NC; ++n)
        acc += sc[n] * pr[n] + ssh[n] * pi[n];

    float silu = acc / (1.0f + expf(-acc));
    x[bs * DC + d] += attn[bs * DC + d] + res_scale[l] * silu;
}

// ---------------------------------------------------------------------------
// Kernel 4a: pre-split x into packed bf16 hi/lo A-fragments in WMMA lane
// layout. Fragment (mt,kc,lane) = 16 bf16 at offset ((mt*4+kc)*32+lane)*16.
// ---------------------------------------------------------------------------
__global__ void pack_a_kernel(const float* __restrict__ x,
                              __bf16* __restrict__ a_hi,
                              __bf16* __restrict__ a_lo) {
    int t = blockIdx.x * blockDim.x + threadIdx.x;   // 0..4095
    if (t >= 32 * 4 * 32) return;
    int lane = t & 31;
    int kc   = (t >> 5) & 3;
    int mt   = t >> 7;
    int row  = mt * 16 + (lane & 15);
    int half = lane >> 4;
    const float* xr = x + (size_t)row * DC;
    size_t base = (size_t)((mt * 4 + kc) * 32 + lane) * 16;
    #pragma unroll
    for (int j = 0; j < 16; ++j) {
        int K = kc * 32 + half * 16 + j;
        float av = xr[K];
        __bf16 h = (__bf16)av;
        a_hi[base + j] = h;
        a_lo[base + j] = (__bf16)(av - (float)h);
    }
}

// ---------------------------------------------------------------------------
// Kernel 4b: logits = x @ out_proj^T via bf16x3 split WMMA.
// One wave per N-tile: B tile loaded & split once into registers,
// then sweep all 32 M-tiles. Loads divergence-free; stores guarded and
// non-temporal (output is write-once, 103 MB).
// ---------------------------------------------------------------------------
__global__ void logits_kernel(const __bf16* __restrict__ a_hi,
                              const __bf16* __restrict__ a_lo,
                              const float* __restrict__ w,
                              float* __restrict__ out) {
    const int NT = (VC + 15) / 16;                       // 3142 N-tiles
    int wave = (blockIdx.x * blockDim.x + threadIdx.x) >> 5;  // wave-uniform
    int lane = threadIdx.x & 31;
    if (wave >= NT) return;                              // wave-uniform exit

    int nt   = wave;
    int col  = nt * 16 + (lane & 15);
    int half = lane >> 4;
    bool colv = (col < VC);
    int  colc = colv ? col : (VC - 1);                   // clamp loads, not stores
    const float* wr = w + (size_t)colc * DC;

    // Load + split the B column tile once: K = kc*32 + half*16 + j
    v16bf bhi[4], blo[4];
    #pragma unroll
    for (int kc = 0; kc < 4; ++kc) {
        const float* wp = wr + kc * 32 + half * 16;      // 16 contiguous floats
        #pragma unroll
        for (int j = 0; j < 16; ++j) {
            float bv = wp[j];
            __bf16 h = (__bf16)bv;
            bhi[kc][j] = h;
            blo[kc][j] = (__bf16)(bv - (float)h);
        }
    }

    // Sweep all M-tiles with pre-packed A fragments (pure b128 loads).
    for (int mt = 0; mt < 32; ++mt) {
        v8f acc = {};
        #pragma unroll
        for (int kc = 0; kc < 4; ++kc) {
            size_t base = (size_t)((mt * 4 + kc) * 32 + lane) * 16;
            v16bf ah = *(const v16bf*)(a_hi + base);
            v16bf al = *(const v16bf*)(a_lo + base);
            acc = __builtin_amdgcn_wmma_f32_16x16x32_bf16(false, ah, false, bhi[kc],
                                                          (short)0, acc, false, false);
            acc = __builtin_amdgcn_wmma_f32_16x16x32_bf16(false, ah, false, blo[kc],
                                                          (short)0, acc, false, false);
            acc = __builtin_amdgcn_wmma_f32_16x16x32_bf16(false, al, false, bhi[kc],
                                                          (short)0, acc, false, false);
        }
        if (colv) {
            #pragma unroll
            for (int r = 0; r < 8; ++r) {
                int m = mt * 16 + r + half * 8;          // C/D layout: VGPR r -> M=r(+8)
                __builtin_nontemporal_store(acc[r], &out[(size_t)m * VC + col]);
            }
        }
    }
}

// ---------------------------------------------------------------------------
// Host launch
// ---------------------------------------------------------------------------
extern "C" void kernel_launch(void* const* d_in, const int* in_sizes, int n_in,
                              void* d_out, int out_size, void* d_ws, size_t ws_size,
                              hipStream_t stream) {
    const int*   tokens     = (const int*)  d_in[0];
    const float* emb        = (const float*)d_in[1];
    const float* wq         = (const float*)d_in[2];
    const float* bq         = (const float*)d_in[3];
    const float* wk         = (const float*)d_in[4];
    const float* bk         = (const float*)d_in[5];
    const float* w_out      = (const float*)d_in[6];
    const float* b_out      = (const float*)d_in[7];
    const float* out_scale  = (const float*)d_in[8];
    const float* Wf         = (const float*)d_in[9];
    const float* Bf         = (const float*)d_in[10];
    const float* proj_real  = (const float*)d_in[11];
    const float* proj_imag  = (const float*)d_in[12];
    const float* attn_scale = (const float*)d_in[13];
    const float* res_scale  = (const float*)d_in[14];
    const float* out_proj   = (const float*)d_in[15];
    float* out = (float*)d_out;

    // Workspace layout: floats [states | x | attn | cos_sum | sin_sum | invw],
    // then packed bf16 A fragments [a_hi | a_lo].
    float* states = (float*)d_ws;
    float* xbuf   = states + BC * SC * DC;      // 65536 floats
    float* attn   = xbuf   + BC * SC * DC;      // 65536
    float* csum   = attn   + BC * SC * DC;      // 65536
    float* ssum   = csum   + BC * SC * NC;      // 262144
    float* invw   = ssum   + BC * SC * NC;      // 262144
    __bf16* apack_hi = (__bf16*)(invw + NC * DC);           // 65536 bf16
    __bf16* apack_lo = apack_hi + BC * SC * DC;             // 65536 bf16
    // total ≈ 3.4 MB

    scan_kernel<<<1, 256, 0, stream>>>(tokens, emb, states, xbuf);

    for (int l = 0; l < LC; ++l) {
        attn_kernel<<<BC * SC, 256, 0, stream>>>(xbuf, states, wq, bq, wk, bk,
                                                 w_out, b_out, out_scale,
                                                 attn_scale, attn, l);
        invw_kernel<<<(NC * DC + 255) / 256, 256, 0, stream>>>(Wf, invw, l);
        fourier_kernel<<<BC * SC, 256, 0, stream>>>(xbuf, invw, Bf, csum, ssum, l);
        res_kernel<<<BC * SC, DC, 0, stream>>>(xbuf, attn, csum, ssum,
                                               proj_real, proj_imag, res_scale, l);
    }

    pack_a_kernel<<<16, 256, 0, stream>>>(xbuf, apack_hi, apack_lo);

    const int NT = (VC + 15) / 16;             // 3142 waves, 8 per block
    int blocks = (NT + 7) / 8;                 // 393 blocks
    logits_kernel<<<blocks, 256, 0, stream>>>(apack_hi, apack_lo, out_proj, out);
}
